// AttentionBlock_28200755265922
// MI455X (gfx1250) — compile-verified, matrix operand order checked by reading, not run on hardware
//
#include <hip/hip_runtime.h>

#define B_ 16
#define C_ 512
#define L_ 1024
#define GROUPS 32
#define CPG 16
#define CH 64
#define EPS_ 1e-4f
#define QK_SCALE 0.35355339059327373f   // 64^-0.25

typedef __attribute__((ext_vector_type(16))) __bf16 bf16x16;
typedef __attribute__((ext_vector_type(16))) unsigned short u16x16;
typedef __attribute__((ext_vector_type(4)))  unsigned int   u32x4;
typedef __attribute__((ext_vector_type(4)))  float          f32x4;
typedef __attribute__((ext_vector_type(8)))  float          f32x8;

union FragU { u16x16 u; u32x4 q[2]; bf16x16 b; };

__device__ __forceinline__ unsigned short f2bf(float f) {
  union { float f; unsigned u; } x; x.f = f;
  unsigned r = x.u + 0x7FFFu + ((x.u >> 16) & 1u);   // round-to-nearest-even
  return (unsigned short)(r >> 16);
}

// Inverse of the 16-bit WMMA A/B K-interleave (ISA 7.12.2):
// given K index k in [0,32), which half h (lane>=16?) and frag element e holds it.
__device__ __forceinline__ void inv_kpat(int k, int& h, int& e) {
  if (k < 16) { h = k >> 3; e = k & 7; }
  else        { h = (k - 16) >> 3; e = 8 + ((k - 16) & 7); }
}

__device__ __forceinline__ f32x8 wmma_bf16(const FragU& a, const FragU& b, f32x8 c) {
  return __builtin_amdgcn_wmma_f32_16x16x32_bf16(false, a.b, false, b.b,
                                                 (short)0, c, false, false);
}

// ---------------- kernel 0: fp32 -> bf16 weight convert (x4) ---------------
__global__ void cvt_bf16_kernel(const float* __restrict__ src,
                                unsigned int* __restrict__ dst, int n4) {
  int i = blockIdx.x * 256 + threadIdx.x;
  if (i < n4) {
    f32x4 v = *(const f32x4*)(src + i * 4);
    dst[i * 2]     = (unsigned)f2bf(v.x) | ((unsigned)f2bf(v.y) << 16);
    dst[i * 2 + 1] = (unsigned)f2bf(v.z) | ((unsigned)f2bf(v.w) << 16);
  }
}

// ---------------- kernel 1: GroupNorm -> bf16 ------------------------------
__global__ void __launch_bounds__(256) groupnorm_kernel(
    const float* __restrict__ x, const float* __restrict__ gw,
    const float* __restrict__ gb, unsigned short* __restrict__ xn) {
  int bg = blockIdx.x;                // b*GROUPS + g
  int b = bg >> 5, g = bg & 31;
  const float* xp = x + ((size_t)(b * C_ + g * CPG)) * L_;
  unsigned short* xo = xn + ((size_t)(b * C_ + g * CPG)) * L_;
  int tid = threadIdx.x;
  const f32x4* xp4 = (const f32x4*)xp;
  float s = 0.f, ss = 0.f;
  for (int i = tid; i < CPG * L_ / 4; i += 256) {
    f32x4 v = xp4[i];
    s  += v.x + v.y + v.z + v.w;
    ss += v.x * v.x + v.y * v.y + v.z * v.z + v.w * v.w;
  }
  __shared__ float rs[256], rss[256];
  rs[tid] = s; rss[tid] = ss;
  __syncthreads();
  for (int off = 128; off > 0; off >>= 1) {
    if (tid < off) { rs[tid] += rs[tid + off]; rss[tid] += rss[tid + off]; }
    __syncthreads();
  }
  float mean = rs[0] * (1.f / (CPG * L_));
  float var  = rss[0] * (1.f / (CPG * L_)) - mean * mean;
  float rstd = rsqrtf(var + EPS_);
  unsigned* xo32 = (unsigned*)xo;
  for (int i = tid; i < CPG * L_ / 4; i += 256) {
    int c = g * CPG + (i >> 8);            // 256 float4 per channel
    float wc = gw[c] * rstd, bc = gb[c] - mean * wc;
    f32x4 v = xp4[i];
    xo32[i * 2]     = (unsigned)f2bf(v.x * wc + bc) | ((unsigned)f2bf(v.y * wc + bc) << 16);
    xo32[i * 2 + 1] = (unsigned)f2bf(v.z * wc + bc) | ((unsigned)f2bf(v.w * wc + bc) << 16);
  }
}

// ---------------- kernel 2: qkv GEMM + bias + scale ------------------------
__global__ void __launch_bounds__(256) qkv_gemm_kernel(
    const unsigned short* __restrict__ W,    // [1536][512] bf16
    const unsigned short* __restrict__ Xn,   // [B][512][1024] bf16
    const float* __restrict__ bias,          // [1536]
    unsigned short* __restrict__ Qb, unsigned short* __restrict__ Kb,
    unsigned short* __restrict__ Vb) {
  int b = blockIdx.z;
  int oBase = blockIdx.x * 128;
  int lBase = blockIdx.y * 16;
  int tid = threadIdx.x;
  int wave = tid >> 5, lane = tid & 31;
  int h = lane >> 4, mr = lane & 15;
  __shared__ u16x16 sBf[2][32];              // frag-major B tile, double buffered
  const unsigned short* xptr = Xn + (size_t)b * C_ * L_;
  const unsigned short* wrow = W + (size_t)(oBase + wave * 16 + mr) * C_;

  auto stage = [&](int buf, int k0) {        // one u32 per thread
    int kk = tid >> 3, nc = tid & 7;
    unsigned d = *(const unsigned*)(xptr + (size_t)(k0 + kk) * L_ + lBase + nc * 2);
    int hh, ee; inv_kpat(kk, hh, ee);
    unsigned short* base = (unsigned short*)&sBf[buf][0];
    int l0 = (nc * 2 + 16 * hh) * 16 + ee;   // lane = n + 16h ; next n -> +16
    base[l0]      = (unsigned short)d;
    base[l0 + 16] = (unsigned short)(d >> 16);
  };

  stage(0, 0);
  __syncthreads();
  f32x8 acc = {0.f, 0.f, 0.f, 0.f, 0.f, 0.f, 0.f, 0.f};
  int buf = 0;
  for (int k0 = 0; k0 < C_; k0 += 32) {
    if (k0 + 32 < C_) {
      stage(buf ^ 1, k0 + 32);
      __builtin_prefetch(wrow + k0 + 64, 0, 1);
    }
    FragU fa, fb;
    fa.q[0] = *(const u32x4*)(wrow + k0 + 8 * h);        // K = 8h..8h+7
    fa.q[1] = *(const u32x4*)(wrow + k0 + 16 + 8 * h);   // K = 16+8h..+7
    fb.u = sBf[buf][lane];
    acc = wmma_bf16(fa, fb, acc);
    __syncthreads();
    buf ^= 1;
  }
  // uniform routing: entire 128-row block lies in one of q/k/v
  int sec = oBase >> 9;
  unsigned short* dst = (sec == 0) ? Qb : (sec == 1) ? Kb : Vb;
  float scl = (sec < 2) ? QK_SCALE : 1.f;
  int l = lBase + mr;
#pragma unroll
  for (int r = 0; r < 8; ++r) {
    int o = oBase + wave * 16 + r + 8 * h;   // C/D: m = r + 8h, n = lane&15
    float val = (acc[r] + bias[o]) * scl;
    dst[((size_t)b * C_ + (o & 511)) * L_ + l] = f2bf(val);
  }
}

// ---------------- kernel 3: flash attention --------------------------------
__global__ void __launch_bounds__(256) attn_kernel(
    const unsigned short* __restrict__ Qb,
    const unsigned short* __restrict__ Kb,
    const unsigned short* __restrict__ Vb,
    unsigned short* __restrict__ Ab) {
  int bh = blockIdx.y;
  int tid = threadIdx.x;
  int w = tid >> 5, lane = tid & 31;
  int h = lane >> 4, mr = lane & 15;
  int tBase = blockIdx.x * 128 + w * 16;

  __shared__ u16x16 sKf[2][2][2][32];   // [buf][chChunk][sHalf][lane]
  __shared__ u16x16 sVf[2][4][32];      // [buf][chTile][lane]
  __shared__ u16x16 sQf[8][2][32];      // [wave][chChunk][lane]
  __shared__ u16x16 sPf[8][32];         // [wave][lane]
  __shared__ float sM[8][16], sL[8][16], sF[8][16];

  const unsigned short* qp = Qb + (size_t)bh * CH * L_;
  const unsigned short* kp = Kb + (size_t)bh * CH * L_;
  const unsigned short* vp = Vb + (size_t)bh * CH * L_;

  // stage this wave's Q tile in frag-major order (A-frag: m=t, k=ch)
#pragma unroll
  for (int i = 0; i < 16; ++i) {
    int idx = lane + i * 32;           // (c, t-pair)
    int c = idx >> 3, tc = idx & 7, t0 = tc * 2;
    unsigned d = *(const unsigned*)(qp + (size_t)c * L_ + tBase + t0);
    int chunk = c >> 5, kk = c & 31;
    int hh, ee; inv_kpat(kk, hh, ee);
    unsigned short* base = (unsigned short*)&sQf[w][chunk][0];
    int l0 = (t0 + 16 * hh) * 16 + ee;
    base[l0]      = (unsigned short)d;
    base[l0 + 16] = (unsigned short)(d >> 16);
  }
  if (lane < 16) { sM[w][lane] = -3.0e38f; sL[w][lane] = 0.f; }

  auto stageKV = [&](int buf, int sBase) {
#pragma unroll
    for (int i = 0; i < 4; ++i) {
      int idx = tid + i * 256;         // 1024 dwords cover 64x32 per tensor
      int c = idx >> 4, sc = idx & 15, s0 = sc * 2;
      unsigned dk = *(const unsigned*)(kp + (size_t)c * L_ + sBase + s0);
      unsigned dv = *(const unsigned*)(vp + (size_t)c * L_ + sBase + s0);
      // K -> B-frag (k=ch, n=s): lane = (s&15) + 16h(ch)
      int chunk = c >> 5, kk = c & 31;
      int hh, ee; inv_kpat(kk, hh, ee);
      int half = s0 >> 4, mrr = s0 & 15;
      unsigned short* kb = (unsigned short*)&sKf[buf][chunk][half][0];
      int l0 = (mrr + 16 * hh) * 16 + ee;
      kb[l0]      = (unsigned short)dk;
      kb[l0 + 16] = (unsigned short)(dk >> 16);   // s0+1 -> next lane
      // V -> A-frag (m=ch, k=s): both halves of dword land in same lane,
      // consecutive e -> single b32 store
      int ct = c >> 4, ml = c & 15;
      int hv, ev; inv_kpat(s0, hv, ev);
      unsigned* vd = (unsigned*)&sVf[buf][ct][ml + 16 * hv];
      vd[ev >> 1] = dv;
    }
  };

  stageKV(0, 0);
  __syncthreads();

  FragU aq0, aq1;
  aq0.u = sQf[w][0][lane];
  aq1.u = sQf[w][1][lane];

  f32x8 accA[4];
#pragma unroll
  for (int ct = 0; ct < 4; ++ct)
#pragma unroll
    for (int r = 0; r < 8; ++r) accA[ct][r] = 0.f;

  int buf = 0;
  for (int sBase = 0; sBase < L_; sBase += 32) {
    if (sBase + 32 < L_) {
      stageKV(buf ^ 1, sBase + 32);
      __builtin_prefetch(kp + sBase + 64, 0, 1);
      __builtin_prefetch(vp + sBase + 64, 0, 1);
    }
    f32x8 sc0 = {0.f,0.f,0.f,0.f,0.f,0.f,0.f,0.f};
    f32x8 sc1 = {0.f,0.f,0.f,0.f,0.f,0.f,0.f,0.f};
    FragU bk;
    bk.u = sKf[buf][0][0][lane]; sc0 = wmma_bf16(aq0, bk, sc0);
    bk.u = sKf[buf][1][0][lane]; sc0 = wmma_bf16(aq1, bk, sc0);
    bk.u = sKf[buf][0][1][lane]; sc1 = wmma_bf16(aq0, bk, sc1);
    bk.u = sKf[buf][1][1][lane]; sc1 = wmma_bf16(aq1, bk, sc1);

    // online softmax: row t = r + 8h lives on the 16 lanes of this half
    unsigned short* pb = (unsigned short*)&sPf[w][0];
    int hp = mr >> 3, ep = mr & 7;     // frag position of column s=mr / s=16+mr
#pragma unroll
    for (int r = 0; r < 8; ++r) {
      int row = r + 8 * h;
      float rmax = fmaxf(sc0[r], sc1[r]);
#pragma unroll
      for (int off = 1; off < 16; off <<= 1)
        rmax = fmaxf(rmax, __shfl_xor(rmax, off, 32));
      float mold = sM[w][row];
      float mnew = fmaxf(mold, rmax);
      float fac  = __expf(mold - mnew);
      float p0 = __expf(sc0[r] - mnew);
      float p1 = __expf(sc1[r] - mnew);
      float rsum = p0 + p1;
#pragma unroll
      for (int off = 1; off < 16; off <<= 1)
        rsum += __shfl_xor(rsum, off, 32);
      // P^T B-frag (k=s, n=t): lane = t + 16*(mr>>3); elements ep / ep+8
      int l0 = (row + 16 * hp) * 16 + ep;
      pb[l0]     = f2bf(p0);
      pb[l0 + 8] = f2bf(p1);
      if (mr == 0) {
        sM[w][row] = mnew;
        sL[w][row] = sL[w][row] * fac + rsum;
        sF[w][row] = fac;
      }
    }

    float fac_t = sF[w][mr];           // accumulator n-index (t) == lane&15
#pragma unroll
    for (int ct = 0; ct < 4; ++ct)
#pragma unroll
      for (int r = 0; r < 8; ++r) accA[ct][r] *= fac_t;

    FragU bp;
    bp.u = sPf[w][lane];
#pragma unroll
    for (int ct = 0; ct < 4; ++ct) {
      FragU av;
      av.u = sVf[buf][ct][lane];
      accA[ct] = wmma_bf16(av, bp, accA[ct]);
    }
    __syncthreads();
    buf ^= 1;
  }

  float inv = 1.0f / sL[w][mr];
#pragma unroll
  for (int ct = 0; ct < 4; ++ct)
#pragma unroll
    for (int r = 0; r < 8; ++r) {
      int c = ct * 16 + r + 8 * h;
      Ab[((size_t)bh * CH + c) * L_ + tBase + mr] = f2bf(accA[ct][r] * inv);
    }
}

// ---------------- kernel 4: proj GEMM + bias + residual --------------------
__global__ void __launch_bounds__(256) proj_gemm_kernel(
    const unsigned short* __restrict__ W,    // [512][512] bf16
    const unsigned short* __restrict__ Ab,   // [B][512][1024] bf16
    const float* __restrict__ bias, const float* __restrict__ x,
    float* __restrict__ out) {
  int b = blockIdx.z;
  int oBase = blockIdx.x * 128;
  int lBase = blockIdx.y * 16;
  int tid = threadIdx.x;
  int wave = tid >> 5, lane = tid & 31;
  int h = lane >> 4, mr = lane & 15;
  __shared__ u16x16 sBf[2][32];
  const unsigned short* aptr = Ab + (size_t)b * C_ * L_;
  const unsigned short* wrow = W + (size_t)(oBase + wave * 16 + mr) * C_;

  auto stage = [&](int buf, int k0) {
    int kk = tid >> 3, nc = tid & 7;
    unsigned d = *(const unsigned*)(aptr + (size_t)(k0 + kk) * L_ + lBase + nc * 2);
    int hh, ee; inv_kpat(kk, hh, ee);
    unsigned short* base = (unsigned short*)&sBf[buf][0];
    int l0 = (nc * 2 + 16 * hh) * 16 + ee;
    base[l0]      = (unsigned short)d;
    base[l0 + 16] = (unsigned short)(d >> 16);
  };

  stage(0, 0);
  __syncthreads();
  f32x8 acc = {0.f, 0.f, 0.f, 0.f, 0.f, 0.f, 0.f, 0.f};
  int buf = 0;
  for (int k0 = 0; k0 < C_; k0 += 32) {
    if (k0 + 32 < C_) {
      stage(buf ^ 1, k0 + 32);
      __builtin_prefetch(wrow + k0 + 64, 0, 1);
    }
    FragU fa, fb;
    fa.q[0] = *(const u32x4*)(wrow + k0 + 8 * h);
    fa.q[1] = *(const u32x4*)(wrow + k0 + 16 + 8 * h);
    fb.u = sBf[buf][lane];
    acc = wmma_bf16(fa, fb, acc);
    __syncthreads();
    buf ^= 1;
  }
  int l = lBase + mr;
#pragma unroll
  for (int r = 0; r < 8; ++r) {
    int o = oBase + wave * 16 + r + 8 * h;
    size_t di = ((size_t)b * C_ + o) * L_ + l;
    out[di] = x[di] + acc[r] + bias[o];
  }
}

// ---------------- launcher -------------------------------------------------
extern "C" void kernel_launch(void* const* d_in, const int* in_sizes, int n_in,
                              void* d_out, int out_size, void* d_ws, size_t ws_size,
                              hipStream_t stream) {
  const float* x      = (const float*)d_in[0];
  const float* norm_w = (const float*)d_in[1];
  const float* norm_b = (const float*)d_in[2];
  const float* qkv_w  = (const float*)d_in[3];
  const float* qkv_b  = (const float*)d_in[4];
  const float* proj_w = (const float*)d_in[5];
  const float* proj_b = (const float*)d_in[6];
  float* out = (float*)d_out;

  char* ws = (char*)d_ws;
  const size_t SZ = (size_t)B_ * C_ * L_ * sizeof(unsigned short);  // 16 MiB
  unsigned short* xn = (unsigned short*)(ws);
  unsigned short* qb = (unsigned short*)(ws + SZ);
  unsigned short* kb = (unsigned short*)(ws + 2 * SZ);
  unsigned short* vb = (unsigned short*)(ws + 3 * SZ);
  unsigned short* ab = (unsigned short*)(ws + 4 * SZ);
  unsigned short* wq = (unsigned short*)(ws + 5 * SZ);
  unsigned short* wp = (unsigned short*)(ws + 5 * SZ +
                                         (size_t)3 * C_ * C_ * sizeof(unsigned short));
  (void)ws_size; (void)in_sizes; (void)n_in; (void)out_size;

  int nq4 = 3 * C_ * C_ / 4, np4 = C_ * C_ / 4;
  cvt_bf16_kernel<<<(nq4 + 255) / 256, 256, 0, stream>>>(qkv_w, (unsigned*)wq, nq4);
  cvt_bf16_kernel<<<(np4 + 255) / 256, 256, 0, stream>>>(proj_w, (unsigned*)wp, np4);
  groupnorm_kernel<<<B_ * GROUPS, 256, 0, stream>>>(x, norm_w, norm_b, xn);
  qkv_gemm_kernel<<<dim3(12, 64, 16), 256, 0, stream>>>(wq, xn, qkv_b, qb, kb, vb);
  attn_kernel<<<dim3(8, 128), 256, 0, stream>>>(qb, kb, vb, ab);
  proj_gemm_kernel<<<dim3(4, 64, 16), 256, 0, stream>>>(wp, ab, proj_b, x, out);
}
